// ImprovedPhysicsInformedFNO_4982162063395
// MI455X (gfx1250) — matrix operand order repaired and checked by reading, not compiled, not run on hardware
//
#include <hip/hip_runtime.h>
#include <math.h>

// ---------------- Problem constants ----------------
#define B_    2048
#define C_    128
#define T_    300
#define M_    24
#define L_    6
#define NP_   5
#define KF_   48          // 24 real + 24 imag spectral coeffs
#define KFP_  64          // padded K for irfft GEMM
#define TPAD_ 320         // padded K for forward DFT GEMM
#define N3_   (C_*T_)     // 38400

typedef __attribute__((ext_vector_type(16))) __bf16 v16bf;
typedef __attribute__((ext_vector_type(8)))  float  v8f;

union BF16x16 { v16bf v; __bf16 h[16]; };

__device__ inline v8f vzero() {
  v8f z;
#pragma unroll
  for (int i = 0; i < 8; i++) z[i] = 0.0f;
  return z;
}

__device__ inline float gelu_f(float x) {
  return 0.5f * x * (1.0f + erff(x * 0.70710678118654752f));
}

__device__ inline v8f bwmma(v16bf a, v16bf b, v8f c) {
  // v_wmma_f32_16x16x32_bf16 : D = A(16x32) * B(32x16) + C
  return __builtin_amdgcn_wmma_f32_16x16x32_bf16(false, a, false, b, (short)0, c, false, false);
}

// ---- 16B-granular data movement helpers ----

// Convert 8 consecutive f32 (16B aligned) to 8 bf16, store 16B to LDS.
__device__ inline void cvt_store8(__bf16* dst, const float* src) {
  float4 a = *(const float4*)src;
  float4 b = *(const float4*)(src + 4);
  union { uint4 u; __bf16 h[8]; } o;
  o.h[0] = (__bf16)a.x; o.h[1] = (__bf16)a.y; o.h[2] = (__bf16)a.z; o.h[3] = (__bf16)a.w;
  o.h[4] = (__bf16)b.x; o.h[5] = (__bf16)b.y; o.h[6] = (__bf16)b.z; o.h[7] = (__bf16)b.w;
  *(uint4*)dst = o.u;
}

__device__ inline void zero16(__bf16* dst) {
  uint4 z; z.x = 0; z.y = 0; z.z = 0; z.w = 0;
  *(uint4*)dst = z;
}

// CDNA5 async global->LDS copy of 16 bytes per lane (ASYNCcnt-tracked).
// LDS byte offset = low 32 bits of the generic pointer (flat LDS aperture mapping).
__device__ inline void copy16_async(__bf16* ldsDst, const __bf16* gsrc) {
  unsigned loff = (unsigned)(unsigned long long)(uintptr_t)ldsDst;
  unsigned long long ga = (unsigned long long)(uintptr_t)gsrc;
  asm volatile("global_load_async_to_lds_b128 %0, %1, off"
               :: "v"(loff), "v"(ga)
               : "memory");
}
__device__ inline void wait_async0() {
  asm volatile("s_wait_asynccnt 0x0" ::: "memory");
}

// A fragment: 16(M) x 32(K) bf16 tile, row-major in LDS with row stride ldk.
// Lane r (0..15): K 0..7 and 16..23 of row r; lane r+16: K 8..15 and 24..31.
__device__ inline v16bf frag_a(const __bf16* s, int ldk) {
  int lane = threadIdx.x & 31;
  const __bf16* p = s + (lane & 15) * ldk + ((lane >> 4) & 1) * 8;
  BF16x16 o;
#pragma unroll
  for (int i = 0; i < 8; i++) { o.h[i] = p[i]; o.h[8 + i] = p[16 + i]; }
  return o.v;
}

// B fragment: operand stored as B^T tile 16(N) x 32(K) row-major in LDS.
// Lane n (0..15): K 0..7 and 16..23 of column n; lane n+16: K 8..15 and 24..31.
__device__ inline v16bf frag_b(const __bf16* s, int ldk) {
  int lane = threadIdx.x & 31;
  const __bf16* p = s + (lane & 15) * ldk + ((lane >> 4) & 1) * 8;
  BF16x16 o;
#pragma unroll
  for (int i = 0; i < 8; i++) { o.h[i] = p[i]; o.h[8 + i] = p[16 + i]; }
  return o.v;
}

// ---------------- Small conversion / setup kernels ----------------

// Vectorized f32 -> bf16 cast, 8 elements (one 16B LDS-free store) per iteration.
__global__ void k_cast8(const float* __restrict__ s, __bf16* __restrict__ d, long n8) {
  long i = (long)blockIdx.x * 256 + threadIdx.x;
  for (; i < n8; i += (long)gridDim.x * 256) cvt_store8(d + i * 8, s + i * 8);
}

// spec_wr/spec_wi: [L][i][o][m] -> WR/WI/WN (=-WI) : [L][m][o][i] bf16 (B^T layout)
__global__ void k_specw(const float* __restrict__ wr, const float* __restrict__ wi,
                        __bf16* __restrict__ WR, __bf16* __restrict__ WI,
                        __bf16* __restrict__ WN, long n) {
  long idx = (long)blockIdx.x * 256 + threadIdx.x;
  for (; idx < n; idx += (long)gridDim.x * 256) {
    long t = idx;
    int i = (int)(t % C_); t /= C_;
    int o = (int)(t % C_); t /= C_;
    int m = (int)(t % M_); t /= M_;
    int l = (int)t;
    size_t src = (((size_t)l * C_ + i) * C_ + o) * M_ + m;
    float r = wr[src], im = wi[src];
    WR[idx] = (__bf16)r;
    WI[idx] = (__bf16)im;
    WN[idx] = (__bf16)(-im);
  }
}

// Forward DFT basis (B^T): FWDB[n][t], n<24: cos(2pi n t/300), 24<=n<48: -sin, else 0.
// Inverse basis (B^T):     INVB[t][j], j=2m: Gc, j=2m+1: Gs (irfft with Im X0 dropped).
__global__ void k_basis(__bf16* __restrict__ FWDB, __bf16* __restrict__ INVB) {
  int idx = blockIdx.x * 256 + threadIdx.x;
  const float w0 = 6.28318530717958647692f / (float)T_;
  if (idx < KFP_ * TPAD_) {
    int n = idx / TPAD_, t = idx % TPAD_;
    float v = 0.0f;
    if (t < T_) {
      if (n < M_)            v = cosf(w0 * (float)n * (float)t);
      else if (n < 2 * M_)   v = -sinf(w0 * (float)(n - M_) * (float)t);
    }
    FWDB[idx] = (__bf16)v;
  }
  int idx2 = idx - KFP_ * TPAD_;
  if (idx2 >= 0 && idx2 < T_ * KFP_) {
    int t = idx2 / KFP_, j = idx2 % KFP_;
    float v = 0.0f;
    if (j < KF_) {
      int m = j >> 1;
      if ((j & 1) == 0) v = (m == 0) ? (1.0f / (float)T_)
                                     : (2.0f / (float)T_) * cosf(w0 * (float)m * (float)t);
      else              v = (m == 0) ? 0.0f
                                     : (-2.0f / (float)T_) * sinf(w0 * (float)m * (float)t);
    }
    INVB[idx2] = (__bf16)v;
  }
}

// ---------------- Encoder (small layers, scalar; K=5 then K=128) ----------------
__global__ __launch_bounds__(128) void k_encoder(
    const float* __restrict__ p, const float* __restrict__ w1, const float* __restrict__ b1,
    const float* __restrict__ w2, const float* __restrict__ b2, __bf16* __restrict__ h2out) {
  __shared__ float pv[NP_];
  __shared__ float h1[C_];
  int b = blockIdx.x, j = threadIdx.x;
  if (j < NP_) pv[j] = p[(size_t)b * NP_ + j];
  __syncthreads();
  float acc = b1[j];
#pragma unroll
  for (int k = 0; k < NP_; k++) acc += pv[k] * w1[k * C_ + j];
  h1[j] = gelu_f(acc);
  __syncthreads();
  float acc2 = b2[j];
  for (int k = 0; k < C_; k++) acc2 += h1[k] * w2[k * C_ + j];
  h2out[(size_t)b * C_ + j] = (__bf16)gelu_f(acc2);
}

// ---------------- Big encoder GEMM: X0 = H2[2048,128] @ enc_w3[128,38400] + b3 ----------------
__global__ __launch_bounds__(256) void k_enc3(
    const __bf16* __restrict__ H2, const __bf16* __restrict__ W,
    const float* __restrict__ bias, float* __restrict__ X) {
  __shared__ __bf16 sA[128 * 32];   // A tile [m][k]
  __shared__ __bf16 sB[64 * 32];    // B^T tile [n][k]
  int m0 = blockIdx.x * 128, n0 = blockIdx.y * 64;
  int tid = threadIdx.x, lane = tid & 31, w = tid >> 5;
  int wm = (w & 3) * 32, wn = (w >> 2) * 32;
  v8f acc[2][2];
#pragma unroll
  for (int a = 0; a < 2; a++)
#pragma unroll
    for (int bn = 0; bn < 2; bn++) acc[a][bn] = vzero();

  for (int kk = 0; kk < C_; kk += 32) {
    __syncthreads();
    // A tile: verbatim bf16 copy, async 16B per lane
    for (int ci = tid; ci < (128 * 32) / 8; ci += 256) {
      int off = ci * 8;
      copy16_async(sA + off, H2 + (size_t)(m0 + (off >> 5)) * C_ + kk + (off & 31));
    }
    // B^T tile: 16B global read, transposed scalar LDS stores
    for (int ci = tid; ci < (64 * 32) / 8; ci += 256) {
      int k = ci >> 3, n8 = (ci & 7) * 8;
      union { uint4 u; __bf16 h[8]; } v;
      v.u = *(const uint4*)(W + (size_t)(kk + k) * N3_ + n0 + n8);
#pragma unroll
      for (int j = 0; j < 8; j++) sB[(n8 + j) * 32 + k] = v.h[j];
    }
    if (kk + 32 < C_)
      __builtin_prefetch(&W[(size_t)(kk + 32 + (tid >> 6)) * N3_ + n0 + (tid & 63)], 0, 3);
    wait_async0();
    __syncthreads();
    v16bf a0 = frag_a(sA + wm * 32, 32);
    v16bf a1 = frag_a(sA + (wm + 16) * 32, 32);
    v16bf bb0 = frag_b(sB + wn * 32, 32);
    v16bf bb1 = frag_b(sB + (wn + 16) * 32, 32);
    acc[0][0] = bwmma(a0, bb0, acc[0][0]);
    acc[0][1] = bwmma(a0, bb1, acc[0][1]);
    acc[1][0] = bwmma(a1, bb0, acc[1][0]);
    acc[1][1] = bwmma(a1, bb1, acc[1][1]);
  }
  int dr = ((lane >> 4) & 1) * 8, dc = lane & 15;
#pragma unroll
  for (int mi = 0; mi < 2; mi++)
#pragma unroll
    for (int ni = 0; ni < 2; ni++)
#pragma unroll
      for (int e = 0; e < 8; e++) {
        int row = m0 + wm + mi * 16 + dr + e;
        int col = n0 + wn + ni * 16 + dc;
        X[(size_t)row * N3_ + col] = acc[mi][ni][e] + bias[col];
      }
}

// ---------------- Forward DFT as GEMM: [B*C,300] x FWDB^T -> XR/XI [m][b][c] ----------------
__global__ __launch_bounds__(256) void k_dft(
    const float* __restrict__ X, const __bf16* __restrict__ FWDB,
    float* __restrict__ XR, float* __restrict__ XI) {
  __shared__ __bf16 sF[KFP_ * TPAD_];   // full basis, 40KB
  __shared__ __bf16 sA[128 * 32];
  int tid = threadIdx.x, lane = tid & 31, w = tid >> 5;
  // basis: verbatim bf16, async 16B chunks
  for (int ci = tid; ci < (KFP_ * TPAD_) / 8; ci += 256)
    copy16_async(sF + ci * 8, FWDB + ci * 8);
  wait_async0();
  int r0 = blockIdx.x * 128;
  v8f acc[4];
#pragma unroll
  for (int i = 0; i < 4; i++) acc[i] = vzero();
  __syncthreads();
  for (int ks = 0; ks < 10; ks++) {
    __syncthreads();
    if (ks < 9) {
      for (int ci = tid; ci < (128 * 32) / 8; ci += 256) {
        int r = ci >> 2, c = (ci & 3) * 8;
        cvt_store8(sA + r * 32 + c, X + (size_t)(r0 + r) * T_ + ks * 32 + c);
      }
    } else {
      for (int i = tid; i < 128 * 32; i += 256) {
        int r = i >> 5, c = i & 31;
        int t = 288 + c;
        sA[i] = (t < T_) ? (__bf16)X[(size_t)(r0 + r) * T_ + t] : (__bf16)0.0f;
      }
    }
    __syncthreads();
    v16bf va = frag_a(sA + w * 16 * 32, 32);
#pragma unroll
    for (int nt = 0; nt < 4; nt++) {
      v16bf vb = frag_b(sF + nt * 16 * TPAD_ + ks * 32, TPAD_);
      acc[nt] = bwmma(va, vb, acc[nt]);
    }
  }
  int dr = ((lane >> 4) & 1) * 8, dc = lane & 15;
#pragma unroll
  for (int nt = 0; nt < 4; nt++)
#pragma unroll
    for (int e = 0; e < 8; e++) {
      int n = nt * 16 + dc;
      int row = r0 + w * 16 + dr + e;
      int b = row >> 7, c = row & 127;
      float v = acc[nt][e];
      if (n < M_)          XR[((size_t)n * B_ + b) * C_ + c] = v;
      else if (n < 2 * M_) XI[((size_t)(n - M_) * B_ + b) * C_ + c] = v;
    }
}

// ---------------- Spectral channel mix (complex), one mode per blockIdx.x ----------------
// Yr = XR@WR + XI@(-WI) ; Yi = XR@WI + XI@WR.  Output Y[(b*C+o)*48 + 2m(+1)].
__global__ __launch_bounds__(256) void k_specmix(
    const float* __restrict__ XR, const float* __restrict__ XI,
    const __bf16* __restrict__ WR, const __bf16* __restrict__ WI, const __bf16* __restrict__ WN,
    float* __restrict__ Y) {
  __shared__ __bf16 sAr[64 * 32], sAi[64 * 32];
  __shared__ __bf16 sBr[128 * 32], sBi[128 * 32], sBn[128 * 32];
  int m = blockIdx.x, b0 = blockIdx.y * 64;
  int tid = threadIdx.x, lane = tid & 31, w = tid >> 5;
  int wm = (w & 3) * 16, wn = (w >> 2) * 64;
  v8f aR[4], aI[4];
#pragma unroll
  for (int i = 0; i < 4; i++) { aR[i] = vzero(); aI[i] = vzero(); }
  const size_t xbase = (size_t)m * B_ * C_;
  const size_t wbase = (size_t)m * C_ * C_;
  for (int kk = 0; kk < C_; kk += 32) {
    __syncthreads();
    for (int ci = tid; ci < (64 * 32) / 8; ci += 256) {
      int r = ci >> 2, c = (ci & 3) * 8;
      size_t gi = xbase + (size_t)(b0 + r) * C_ + kk + c;
      cvt_store8(sAr + r * 32 + c, XR + gi);
      cvt_store8(sAi + r * 32 + c, XI + gi);
    }
    for (int ci = tid; ci < (128 * 32) / 8; ci += 256) {
      int off = ci * 8;
      size_t gi = wbase + (size_t)(off >> 5) * C_ + kk + (off & 31);
      copy16_async(sBr + off, WR + gi);
      copy16_async(sBi + off, WI + gi);
      copy16_async(sBn + off, WN + gi);
    }
    wait_async0();
    __syncthreads();
    v16bf var = frag_a(sAr + wm * 32, 32);
    v16bf vai = frag_a(sAi + wm * 32, 32);
#pragma unroll
    for (int nt = 0; nt < 4; nt++) {
      v16bf br = frag_b(sBr + (wn + nt * 16) * 32, 32);
      v16bf bi = frag_b(sBi + (wn + nt * 16) * 32, 32);
      v16bf bn = frag_b(sBn + (wn + nt * 16) * 32, 32);
      aR[nt] = bwmma(var, br, aR[nt]);
      aR[nt] = bwmma(vai, bn, aR[nt]);
      aI[nt] = bwmma(var, bi, aI[nt]);
      aI[nt] = bwmma(vai, br, aI[nt]);
    }
  }
  int dr = ((lane >> 4) & 1) * 8, dc = lane & 15;
#pragma unroll
  for (int nt = 0; nt < 4; nt++)
#pragma unroll
    for (int e = 0; e < 8; e++) {
      int b = b0 + wm + dr + e;
      int o = wn + nt * 16 + dc;
      size_t yb = ((size_t)b * C_ + o) * KF_ + 2 * m;
      Y[yb]     = aR[nt][e];
      Y[yb + 1] = aI[nt][e];
    }
}

// ---------------- Fused: irfft GEMM + 1x1 conv GEMM + bias + GELU + residual ----------------
// One block = one sample b, one 128-wide t chunk. Output [128 o][128 t].
__global__ __launch_bounds__(256) void k_fused(
    const float* __restrict__ Xold, const float* __restrict__ Y,
    const __bf16* __restrict__ INVB, const __bf16* __restrict__ CW,
    const float* __restrict__ cb, float* __restrict__ Xnew) {
  __shared__ __bf16 sY[128 * KFP_];   // Y rows (o), 16KB
  __shared__ __bf16 sG[128 * KFP_];   // inverse basis B^T rows (t), 16KB
  __shared__ __bf16 sW[128 * 32];     // conv weight slice [o][c]
  __shared__ __bf16 sX[128 * 32];     // x^T slice [t][c]
  int b = blockIdx.x, t0 = blockIdx.y * 128;
  int tid = threadIdx.x, lane = tid & 31, w = tid >> 5;
  int wm = (w & 3) * 32, wn = (w >> 2) * 64;
  bool fullT = (t0 + 128) <= T_;

  for (int ci = tid; ci < (128 * KFP_) / 8; ci += 256) {
    int r = ci >> 3, j8 = (ci & 7) * 8;
    __bf16* dy = sY + r * KFP_ + j8;
    if (j8 < KF_) cvt_store8(dy, Y + ((size_t)b * C_ + r) * KF_ + j8);
    else          zero16(dy);
    __bf16* dg = sG + r * KFP_ + j8;
    int t = t0 + r;
    if (t < T_) copy16_async(dg, INVB + t * KFP_ + j8);
    else        zero16(dg);
  }
  v8f acc[2][4];
#pragma unroll
  for (int mi = 0; mi < 2; mi++)
#pragma unroll
    for (int nt = 0; nt < 4; nt++) acc[mi][nt] = vzero();
  wait_async0();
  __syncthreads();

  // GEMM1: x1 = Y @ G^T   (K = 64, 2 steps)
#pragma unroll
  for (int ks = 0; ks < 2; ks++) {
    v16bf va0 = frag_a(sY + wm * KFP_ + ks * 32, KFP_);
    v16bf va1 = frag_a(sY + (wm + 16) * KFP_ + ks * 32, KFP_);
#pragma unroll
    for (int nt = 0; nt < 4; nt++) {
      v16bf vb = frag_b(sG + (wn + nt * 16) * KFP_ + ks * 32, KFP_);
      acc[0][nt] = bwmma(va0, vb, acc[0][nt]);
      acc[1][nt] = bwmma(va1, vb, acc[1][nt]);
    }
  }

  // GEMM2: x2 += W @ x_b  (K = 128, 4 steps)
  for (int kk = 0; kk < C_; kk += 32) {
    __syncthreads();
    for (int ci = tid; ci < (128 * 32) / 8; ci += 256) {
      int off = ci * 8;
      copy16_async(sW + off, CW + (size_t)(off >> 5) * C_ + kk + (off & 31));
    }
    if (fullT) {
      for (int ci = tid; ci < (32 * 128) / 4; ci += 256) {
        int c = ci >> 5, t4 = (ci & 31) * 4;
        float4 v = *(const float4*)(Xold + ((size_t)b * C_ + kk + c) * T_ + t0 + t4);
        sX[(t4 + 0) * 32 + c] = (__bf16)v.x;
        sX[(t4 + 1) * 32 + c] = (__bf16)v.y;
        sX[(t4 + 2) * 32 + c] = (__bf16)v.z;
        sX[(t4 + 3) * 32 + c] = (__bf16)v.w;
      }
    } else {
      for (int i = tid; i < 32 * 128; i += 256) {
        int c = i >> 7, tr = i & 127;
        int t = t0 + tr;
        float v = (t < T_) ? Xold[((size_t)b * C_ + kk + c) * T_ + t] : 0.0f;
        sX[tr * 32 + c] = (__bf16)v;
      }
    }
    if (kk + 32 < C_)
      __builtin_prefetch(&Xold[((size_t)b * C_ + kk + 32 + (tid >> 7)) * T_ + t0 + (tid & 127)], 0, 3);
    wait_async0();
    __syncthreads();
    v16bf va0 = frag_a(sW + wm * 32, 32);
    v16bf va1 = frag_a(sW + (wm + 16) * 32, 32);
#pragma unroll
    for (int nt = 0; nt < 4; nt++) {
      v16bf vb = frag_b(sX + (wn + nt * 16) * 32, 32);
      acc[0][nt] = bwmma(va0, vb, acc[0][nt]);
      acc[1][nt] = bwmma(va1, vb, acc[1][nt]);
    }
  }

  int dr = ((lane >> 4) & 1) * 8, dc = lane & 15;
#pragma unroll
  for (int mi = 0; mi < 2; mi++)
#pragma unroll
    for (int nt = 0; nt < 4; nt++)
#pragma unroll
      for (int e = 0; e < 8; e++) {
        int o = wm + mi * 16 + dr + e;
        int t = t0 + wn + nt * 16 + dc;
        if (t < T_) {
          float v = acc[mi][nt][e] + cb[o];
          size_t ix = ((size_t)b * C_ + o) * T_ + t;
          Xnew[ix] = gelu_f(v) + Xold[ix];
        }
      }
}

// ---------------- Output projection 1: H = gelu(out_w1 @ x + b1), [B][64][300] ----------------
__global__ __launch_bounds__(256) void k_proj1(
    const float* __restrict__ X, const __bf16* __restrict__ OW1,
    const float* __restrict__ ob, float* __restrict__ H) {
  __shared__ __bf16 sW[64 * 32];
  __shared__ __bf16 sX[128 * 32];
  int b = blockIdx.x, t0 = blockIdx.y * 128;
  int tid = threadIdx.x, lane = tid & 31, w = tid >> 5;
  int wm = (w & 1) * 32, wn = (w >> 1) * 32;
  bool fullT = (t0 + 128) <= T_;
  v8f acc[2][2];
#pragma unroll
  for (int a = 0; a < 2; a++)
#pragma unroll
    for (int bn = 0; bn < 2; bn++) acc[a][bn] = vzero();
  for (int kk = 0; kk < C_; kk += 32) {
    __syncthreads();
    for (int ci = tid; ci < (64 * 32) / 8; ci += 256) {
      int off = ci * 8;
      copy16_async(sW + off, OW1 + (size_t)(off >> 5) * C_ + kk + (off & 31));
    }
    if (fullT) {
      for (int ci = tid; ci < (32 * 128) / 4; ci += 256) {
        int c = ci >> 5, t4 = (ci & 31) * 4;
        float4 v = *(const float4*)(X + ((size_t)b * C_ + kk + c) * T_ + t0 + t4);
        sX[(t4 + 0) * 32 + c] = (__bf16)v.x;
        sX[(t4 + 1) * 32 + c] = (__bf16)v.y;
        sX[(t4 + 2) * 32 + c] = (__bf16)v.z;
        sX[(t4 + 3) * 32 + c] = (__bf16)v.w;
      }
    } else {
      for (int i = tid; i < 32 * 128; i += 256) {
        int c = i >> 7, tr = i & 127;
        int t = t0 + tr;
        float v = (t < T_) ? X[((size_t)b * C_ + kk + c) * T_ + t] : 0.0f;
        sX[tr * 32 + c] = (__bf16)v;
      }
    }
    wait_async0();
    __syncthreads();
    v16bf va0 = frag_a(sW + wm * 32, 32);
    v16bf va1 = frag_a(sW + (wm + 16) * 32, 32);
#pragma unroll
    for (int nt = 0; nt < 2; nt++) {
      v16bf vb = frag_b(sX + (wn + nt * 16) * 32, 32);
      acc[0][nt] = bwmma(va0, vb, acc[0][nt]);
      acc[1][nt] = bwmma(va1, vb, acc[1][nt]);
    }
  }
  int dr = ((lane >> 4) & 1) * 8, dc = lane & 15;
#pragma unroll
  for (int mi = 0; mi < 2; mi++)
#pragma unroll
    for (int nt = 0; nt < 2; nt++)
#pragma unroll
      for (int e = 0; e < 8; e++) {
        int o = wm + mi * 16 + dr + e;
        int t = t0 + wn + nt * 16 + dc;
        if (t < T_)
          H[((size_t)b * 64 + o) * T_ + t] = gelu_f(acc[mi][nt][e] + ob[o]);
      }
}

// ---------------- Output projection 2: out[b][t][3] = out_w2 @ H + b2 ----------------
__global__ __launch_bounds__(256) void k_proj2(
    const float* __restrict__ H, const float* __restrict__ w2,
    const float* __restrict__ b2, float* __restrict__ out) {
  __shared__ float sw[3 * 64];
  int tid = threadIdx.x;
  if (tid < 192) sw[tid] = w2[tid];
  __syncthreads();
  int idx = blockIdx.x * 256 + tid;
  if (idx >= B_ * T_) return;
  int b = idx / T_, t = idx % T_;
  float a0 = b2[0], a1 = b2[1], a2 = b2[2];
  for (int c = 0; c < 64; c++) {
    float h = H[((size_t)b * 64 + c) * T_ + t];
    a0 += sw[c] * h;
    a1 += sw[64 + c] * h;
    a2 += sw[128 + c] * h;
  }
  float* o = out + (size_t)idx * 3;
  o[0] = a0; o[1] = a1; o[2] = a2;
}

// ---------------- Launch ----------------
extern "C" void kernel_launch(void* const* d_in, const int* in_sizes, int n_in,
                              void* d_out, int out_size, void* d_ws, size_t ws_size,
                              hipStream_t stream) {
  (void)in_sizes; (void)n_in; (void)out_size; (void)ws_size;
  const float* params = (const float*)d_in[0];
  const float* enc_w1 = (const float*)d_in[1];
  const float* enc_b1 = (const float*)d_in[2];
  const float* enc_w2 = (const float*)d_in[3];
  const float* enc_b2 = (const float*)d_in[4];
  const float* enc_w3 = (const float*)d_in[5];
  const float* enc_b3 = (const float*)d_in[6];
  const float* spec_wr = (const float*)d_in[7];
  const float* spec_wi = (const float*)d_in[8];
  const float* conv_w = (const float*)d_in[9];
  const float* conv_b = (const float*)d_in[10];
  const float* out_w1 = (const float*)d_in[11];
  const float* out_b1 = (const float*)d_in[12];
  const float* out_w2 = (const float*)d_in[13];
  const float* out_b2 = (const float*)d_in[14];

  char* ws = (char*)d_ws;
  size_t off = 0;
  auto take = [&](size_t bytes) -> void* {
    void* p = ws + off;
    off += (bytes + 511) & ~(size_t)511;
    return p;
  };
  float* XA   = (float*)take((size_t)B_ * C_ * T_ * 4);
  float* XB   = (float*)take((size_t)B_ * C_ * T_ * 4);
  float* XRb  = (float*)take((size_t)M_ * B_ * C_ * 4);
  float* XIb  = (float*)take((size_t)M_ * B_ * C_ * 4);
  float* Yb   = (float*)take((size_t)B_ * C_ * KF_ * 4);
  __bf16* H2   = (__bf16*)take((size_t)B_ * C_ * 2);
  __bf16* ENC3 = (__bf16*)take((size_t)C_ * N3_ * 2);
  __bf16* WR   = (__bf16*)take((size_t)L_ * M_ * C_ * C_ * 2);
  __bf16* WI   = (__bf16*)take((size_t)L_ * M_ * C_ * C_ * 2);
  __bf16* WN   = (__bf16*)take((size_t)L_ * M_ * C_ * C_ * 2);
  __bf16* CW   = (__bf16*)take((size_t)L_ * C_ * C_ * 2);
  __bf16* OW1  = (__bf16*)take((size_t)64 * C_ * 2);
  __bf16* FWDB = (__bf16*)take((size_t)KFP_ * TPAD_ * 2);
  __bf16* INVB = (__bf16*)take((size_t)T_ * KFP_ * 2);

  dim3 blk(256);
  k_cast8<<<dim3(1024), blk, 0, stream>>>(enc_w3, ENC3, (long)(C_ * N3_) / 8);
  k_cast8<<<dim3(48), blk, 0, stream>>>(conv_w, CW, (long)(L_ * C_ * C_) / 8);
  k_cast8<<<dim3(4), blk, 0, stream>>>(out_w1, OW1, (long)(64 * C_) / 8);
  k_specw<<<dim3(2048), blk, 0, stream>>>(spec_wr, spec_wi, WR, WI, WN, (long)L_ * M_ * C_ * C_);
  k_basis<<<dim3((KFP_ * TPAD_ + T_ * KFP_ + 255) / 256), blk, 0, stream>>>(FWDB, INVB);

  k_encoder<<<dim3(B_), dim3(128), 0, stream>>>(params, enc_w1, enc_b1, enc_w2, enc_b2, H2);
  k_enc3<<<dim3(B_ / 128, N3_ / 64), blk, 0, stream>>>(H2, ENC3, enc_b3, XA);

  float* cur = XA;
  float* nxt = XB;
  const size_t wstep = (size_t)M_ * C_ * C_;
  for (int l = 0; l < L_; l++) {
    k_dft<<<dim3((B_ * C_) / 128), blk, 0, stream>>>(cur, FWDB, XRb, XIb);
    k_specmix<<<dim3(M_, B_ / 64), blk, 0, stream>>>(XRb, XIb,
        WR + (size_t)l * wstep, WI + (size_t)l * wstep, WN + (size_t)l * wstep, Yb);
    k_fused<<<dim3(B_, 3), blk, 0, stream>>>(cur, Yb, INVB,
        CW + (size_t)l * C_ * C_, conv_b + (size_t)l * C_, nxt);
    float* tmp = cur; cur = nxt; nxt = tmp;
  }

  float* H64 = nxt;  // reuse the free ping-pong buffer (157 MB needed, 315 MB available)
  k_proj1<<<dim3(B_, 3), blk, 0, stream>>>(cur, OW1, out_b1, H64);
  k_proj2<<<dim3((B_ * T_ + 255) / 256), blk, 0, stream>>>(H64, out_w2, out_b2, (float*)d_out);
}
